// SparseMoE_64080912056582
// MI455X (gfx1250) — matrix operand order repaired and checked by reading, not compiled, not run on hardware
//
#include <hip/hip_runtime.h>
#include <hip/hip_bf16.h>
#include <stdint.h>
#include <stddef.h>

#define E_    8
#define D_    1024
#define H_    4096
#define O_    1024
#define N_    4096

typedef __attribute__((ext_vector_type(16))) __bf16 v16bf;
typedef __attribute__((ext_vector_type(8)))  __bf16 v8bf;
typedef __attribute__((ext_vector_type(8)))  float  v8f;

union FragAB { v16bf v; v8bf h[2]; };

#define LDS_STRIDE 40   // bf16 per LDS row: 80B rows -> 16B aligned, conflict-free frag reads

// ---------------- CDNA5 async global->LDS copy (ASYNCcnt path) ----------------
#if defined(__HIP_DEVICE_COMPILE__)
typedef __attribute__((__vector_size__(16))) int b128_t;
typedef __attribute__((address_space(1))) b128_t glb_b128_t;
typedef __attribute__((address_space(3))) b128_t lds_b128_t;

__device__ __forceinline__ void async_copy_b128(const void* g, void* l) {
#if __has_builtin(__builtin_amdgcn_global_load_async_to_lds_b128)
    __builtin_amdgcn_global_load_async_to_lds_b128(
        (glb_b128_t*)(uintptr_t)g, (lds_b128_t*)(uint32_t)(uintptr_t)l, 0, 0);
#else
    uint32_t lo = (uint32_t)(uintptr_t)l;
    asm volatile("global_load_async_to_lds_b128 %0, %1, off"
                 :: "v"(lo), "v"(g) : "memory");
#endif
}
__device__ __forceinline__ void async_wait0() {
#if __has_builtin(__builtin_amdgcn_s_wait_asynccnt)
    __builtin_amdgcn_s_wait_asynccnt(0);
#else
    asm volatile("s_wait_asynccnt 0x0" ::: "memory");
#endif
}
#else
__device__ __forceinline__ void async_copy_b128(const void*, void*) {}
__device__ __forceinline__ void async_wait0() {}
#endif

// ---------------------------------------------------------------- init
__global__ void moe_init_kernel(float* __restrict__ out, int* __restrict__ counts) {
    size_t i = (size_t)blockIdx.x * blockDim.x + threadIdx.x;
    size_t total = (size_t)N_ * O_;
    size_t stride = (size_t)gridDim.x * blockDim.x;
    for (size_t j = i; j < total; j += stride) out[j] = 0.0f;
    if (i < E_) counts[i] = 0;
}

// ---------------------------------------------------------------- x -> bf16
__global__ void moe_cvt_x_kernel(const float* __restrict__ x, __bf16* __restrict__ xb) {
    size_t i = (size_t)blockIdx.x * blockDim.x + threadIdx.x;
    size_t total = (size_t)N_ * D_ / 4;
    if (i >= total) return;
    float4 v = ((const float4*)x)[i];
    __bf16 o[4];
    o[0] = (__bf16)v.x; o[1] = (__bf16)v.y; o[2] = (__bf16)v.z; o[3] = (__bf16)v.w;
    *(uint64_t*)&xb[i * 4] = *(const uint64_t*)o;
}

// ------------------------------------------------- W transpose + f32->bf16
// src: per-expert [R][C] f32 row-major  ->  dst: per-expert [C][R] bf16
__global__ void moe_transpose_cvt_kernel(const float* __restrict__ src,
                                         __bf16* __restrict__ dst, int R, int C) {
    __shared__ float tile[32][33];
    const float*  s = src + (size_t)blockIdx.z * R * C;
    __bf16*       d = dst + (size_t)blockIdx.z * R * C;
    int c0 = blockIdx.x * 32, r0 = blockIdx.y * 32;
    #pragma unroll
    for (int i = threadIdx.y; i < 32; i += 8)
        tile[i][threadIdx.x] = s[(size_t)(r0 + i) * C + c0 + threadIdx.x];
    __syncthreads();
    #pragma unroll
    for (int i = threadIdx.y; i < 32; i += 8)
        d[(size_t)(c0 + i) * R + r0 + threadIdx.x] = (__bf16)tile[threadIdx.x][i];
}

// ---------------------------------------------------------------- gating
__global__ void moe_gate_kernel(const float* __restrict__ x, const float* __restrict__ Wg,
                                const float* __restrict__ bg, int* __restrict__ counts,
                                int* __restrict__ toklist, float* __restrict__ gatew) {
    int gtid = blockIdx.x * blockDim.x + threadIdx.x;
    int tok  = gtid >> 5;
    int lane = threadIdx.x & 31;
    if (tok >= N_) return;
    const float* xr = x + (size_t)tok * D_;
    float acc[E_];
    #pragma unroll
    for (int e = 0; e < E_; e++) acc[e] = 0.0f;
    for (int d = lane; d < D_; d += 32) {
        float xv = xr[d];
        const float4* wg4 = (const float4*)(Wg + (size_t)d * E_);
        float4 w0 = wg4[0], w1 = wg4[1];
        acc[0] += xv * w0.x; acc[1] += xv * w0.y; acc[2] += xv * w0.z; acc[3] += xv * w0.w;
        acc[4] += xv * w1.x; acc[5] += xv * w1.y; acc[6] += xv * w1.z; acc[7] += xv * w1.w;
    }
    #pragma unroll
    for (int e = 0; e < E_; e++) {
        #pragma unroll
        for (int off = 16; off > 0; off >>= 1)
            acc[e] += __shfl_down(acc[e], off, 32);
    }
    if (lane == 0) {
        float lg[E_];
        #pragma unroll
        for (int e = 0; e < E_; e++) lg[e] = acc[e] + bg[e];
        int i0 = 0;
        #pragma unroll
        for (int e = 1; e < E_; e++) if (lg[e] > lg[i0]) i0 = e;
        int i1 = (i0 == 0) ? 1 : 0;
        #pragma unroll
        for (int e = 0; e < E_; e++) if (e != i0 && lg[e] > lg[i1]) i1 = e;
        float m  = lg[i0];
        float e0 = __expf(lg[i0] - m), e1 = __expf(lg[i1] - m);
        float inv = 1.0f / (e0 + e1);
        int p0 = atomicAdd(&counts[i0], 1);
        toklist[i0 * N_ + p0] = tok;  gatew[i0 * N_ + p0] = e0 * inv;
        int p1 = atomicAdd(&counts[i1], 1);
        toklist[i1 * N_ + p1] = tok;  gatew[i1 * N_ + p1] = e1 * inv;
    }
}

// ------------------------------------------- GEMM1: h = relu(xg @ W1 + b1)
// block tile 128x128, K step 32, async double-buffered LDS
// 8 waves: 4(M) x 2(N) -> 32x64 per wave -> 8 WMMA / k-step / wave
__global__ __launch_bounds__(256) void moe_gemm1_kernel(
    const __bf16* __restrict__ xb, const __bf16* __restrict__ W1T,
    const float* __restrict__ b1, const int* __restrict__ counts,
    const int* __restrict__ toklist, __bf16* __restrict__ h, int e) {
    __shared__ __bf16 As[2][128 * LDS_STRIDE];
    __shared__ __bf16 Bs[2][128 * LDS_STRIDE];
    __shared__ int    toks[128];

    const int cnt   = counts[e];
    const int mtile = blockIdx.y;
    if (mtile * 128 >= cnt) return;
    const int n0  = blockIdx.x * 128;
    const int tid = threadIdx.x;

    if (tid < 128) {
        int slot = mtile * 128 + tid;
        toks[tid] = (slot < cnt) ? toklist[e * N_ + slot] : 0;   // clamp: no branches later
    }
    __syncthreads();

    // per-thread staging map: 2 A rows + 2 B rows, one 16B chunk each, fixed all K
    const int part = tid & 3;
    const int r0   = tid >> 2;          // 0..63
    const int r1   = r0 + 64;

    const __bf16* ga[4];
    ga[0] = xb + (size_t)toks[r0] * D_ + part * 8;
    ga[1] = xb + (size_t)toks[r1] * D_ + part * 8;
    const __bf16* Bbase = W1T + ((size_t)e * H_ + n0) * D_ + part * 8;
    ga[2] = Bbase + (size_t)r0 * D_;
    ga[3] = Bbase + (size_t)r1 * D_;

    __bf16* la[2][4];
    #pragma unroll
    for (int b = 0; b < 2; b++) {
        la[b][0] = &As[b][r0 * LDS_STRIDE + part * 8];
        la[b][1] = &As[b][r1 * LDS_STRIDE + part * 8];
        la[b][2] = &Bs[b][r0 * LDS_STRIDE + part * 8];
        la[b][3] = &Bs[b][r1 * LDS_STRIDE + part * 8];
    }

    auto issue_tile = [&](int buf) {
        #pragma unroll
        for (int c = 0; c < 4; c++) {
            async_copy_b128(ga[c], la[buf][c]);
            ga[c] += 32;                 // advance one K-step (64B)
        }
    };

    const int lane = tid & 31;
    const int wave = tid >> 5;
    const int wm = (wave & 3) * 32;
    const int wn = (wave >> 2) * 64;
    const int lo = lane & 15;
    const int hi = lane >> 4;

    v8f acc[2][4] = {};
    issue_tile(0);
    const int nk = D_ / 32;
    for (int kt = 0; kt < nk; kt++) {
        const int cur = kt & 1;
        async_wait0();
        __syncthreads();
        if (kt + 1 < nk) issue_tile(cur ^ 1);

        FragAB a[2], b[4];
        #pragma unroll
        for (int mi = 0; mi < 2; mi++) {
            const __bf16* p = &As[cur][(wm + mi * 16 + lo) * LDS_STRIDE + hi * 8];
            a[mi].h[0] = *(const v8bf*)p;          // K 0..7   (lanes>=16: 8..15)
            a[mi].h[1] = *(const v8bf*)(p + 16);   // K 16..23 (lanes>=16: 24..31)
        }
        #pragma unroll
        for (int ni = 0; ni < 4; ni++) {
            const __bf16* p = &Bs[cur][(wn + ni * 16 + lo) * LDS_STRIDE + hi * 8];
            b[ni].h[0] = *(const v8bf*)p;          // K 0..7 / 16..23 (lanes>=16: 8..15 / 24..31)
            b[ni].h[1] = *(const v8bf*)(p + 16);
        }
        #pragma unroll
        for (int mi = 0; mi < 2; mi++)
            #pragma unroll
            for (int ni = 0; ni < 4; ni++)
                acc[mi][ni] = __builtin_amdgcn_wmma_f32_16x16x32_bf16(
                    false, a[mi].v, false, b[ni].v, (short)0, acc[mi][ni], false, false);
    }

    #pragma unroll
    for (int ni = 0; ni < 4; ni++) {
        int col = n0 + wn + ni * 16 + lo;
        float bias = b1[e * H_ + col];
        #pragma unroll
        for (int mi = 0; mi < 2; mi++) {
            #pragma unroll
            for (int i = 0; i < 8; i++) {
                int slot = mtile * 128 + wm + mi * 16 + hi * 8 + i;
                if (slot < cnt) {
                    float v = acc[mi][ni][i] + bias;
                    h[(size_t)slot * H_ + col] = (__bf16)(v > 0.0f ? v : 0.0f);
                }
            }
        }
    }
}

// ------------------------------- GEMM2: out += g * (h @ W2 + b2), scattered
__global__ __launch_bounds__(256) void moe_gemm2_kernel(
    const __bf16* __restrict__ h, const __bf16* __restrict__ W2T,
    const float* __restrict__ b2, const int* __restrict__ counts,
    const int* __restrict__ toklist, const float* __restrict__ gatew,
    float* __restrict__ out, int e) {
    __shared__ __bf16 As[2][128 * LDS_STRIDE];
    __shared__ __bf16 Bs[2][128 * LDS_STRIDE];

    const int cnt   = counts[e];
    const int mtile = blockIdx.y;
    if (mtile * 128 >= cnt) return;
    const int n0  = blockIdx.x * 128;
    const int tid = threadIdx.x;

    const int part = tid & 3;
    const int r0   = tid >> 2;
    const int r1   = r0 + 64;

    const __bf16* ga[4];
    ga[0] = h + (size_t)(mtile * 128 + r0) * H_ + part * 8;   // rows > cnt: harmless garbage
    ga[1] = h + (size_t)(mtile * 128 + r1) * H_ + part * 8;
    const __bf16* Bbase = W2T + ((size_t)e * O_ + n0) * H_ + part * 8;
    ga[2] = Bbase + (size_t)r0 * H_;
    ga[3] = Bbase + (size_t)r1 * H_;

    __bf16* la[2][4];
    #pragma unroll
    for (int b = 0; b < 2; b++) {
        la[b][0] = &As[b][r0 * LDS_STRIDE + part * 8];
        la[b][1] = &As[b][r1 * LDS_STRIDE + part * 8];
        la[b][2] = &Bs[b][r0 * LDS_STRIDE + part * 8];
        la[b][3] = &Bs[b][r1 * LDS_STRIDE + part * 8];
    }

    auto issue_tile = [&](int buf) {
        #pragma unroll
        for (int c = 0; c < 4; c++) {
            async_copy_b128(ga[c], la[buf][c]);
            ga[c] += 32;
        }
    };

    const int lane = tid & 31;
    const int wave = tid >> 5;
    const int wm = (wave & 3) * 32;
    const int wn = (wave >> 2) * 64;
    const int lo = lane & 15;
    const int hi = lane >> 4;

    v8f acc[2][4] = {};
    issue_tile(0);
    const int nk = H_ / 32;
    for (int kt = 0; kt < nk; kt++) {
        const int cur = kt & 1;
        async_wait0();
        __syncthreads();
        if (kt + 1 < nk) issue_tile(cur ^ 1);

        FragAB a[2], b[4];
        #pragma unroll
        for (int mi = 0; mi < 2; mi++) {
            const __bf16* p = &As[cur][(wm + mi * 16 + lo) * LDS_STRIDE + hi * 8];
            a[mi].h[0] = *(const v8bf*)p;
            a[mi].h[1] = *(const v8bf*)(p + 16);
        }
        #pragma unroll
        for (int ni = 0; ni < 4; ni++) {
            const __bf16* p = &Bs[cur][(wn + ni * 16 + lo) * LDS_STRIDE + hi * 8];
            b[ni].h[0] = *(const v8bf*)p;
            b[ni].h[1] = *(const v8bf*)(p + 16);
        }
        #pragma unroll
        for (int mi = 0; mi < 2; mi++)
            #pragma unroll
            for (int ni = 0; ni < 4; ni++)
                acc[mi][ni] = __builtin_amdgcn_wmma_f32_16x16x32_bf16(
                    false, a[mi].v, false, b[ni].v, (short)0, acc[mi][ni], false, false);
    }

    #pragma unroll
    for (int ni = 0; ni < 4; ni++) {
        int col = n0 + wn + ni * 16 + lo;
        float bias = b2[e * O_ + col];
        #pragma unroll
        for (int mi = 0; mi < 2; mi++) {
            #pragma unroll
            for (int i = 0; i < 8; i++) {
                int slot = mtile * 128 + wm + mi * 16 + hi * 8 + i;
                if (slot < cnt) {
                    int   t = toklist[e * N_ + slot];
                    float g = gatew[e * N_ + slot];
                    float v = (acc[mi][ni][i] + bias) * g;
                    atomicAdd(&out[(size_t)t * O_ + col], v);
                }
            }
        }
    }
}

// ---------------------------------------------------------------- launcher
extern "C" void kernel_launch(void* const* d_in, const int* in_sizes, int n_in,
                              void* d_out, int out_size, void* d_ws, size_t ws_size,
                              hipStream_t stream) {
    const float* x  = (const float*)d_in[0];   // [N,D]
    const float* W1 = (const float*)d_in[1];   // [E,D,H]
    const float* b1 = (const float*)d_in[2];   // [E,H]
    const float* W2 = (const float*)d_in[3];   // [E,H,O]
    const float* b2 = (const float*)d_in[4];   // [E,O]
    const float* Wg = (const float*)d_in[5];   // [D,E]
    const float* bg = (const float*)d_in[6];   // [E]
    float* out = (float*)d_out;

    char* ws = (char*)d_ws;
    __bf16* xb   = (__bf16*)ws;  ws += (size_t)N_ * D_ * 2;
    __bf16* W1T  = (__bf16*)ws;  ws += (size_t)E_ * H_ * D_ * 2;
    __bf16* W2T  = (__bf16*)ws;  ws += (size_t)E_ * O_ * H_ * 2;
    __bf16* hbuf = (__bf16*)ws;  ws += (size_t)N_ * H_ * 2;
    int*    counts  = (int*)ws;  ws += 256;
    int*    toklist = (int*)ws;  ws += (size_t)E_ * N_ * 4;
    float*  gatew   = (float*)ws;

    moe_init_kernel<<<4096, 256, 0, stream>>>(out, counts);
    moe_cvt_x_kernel<<<(N_ * D_ / 4 + 255) / 256, 256, 0, stream>>>(x, xb);
    moe_transpose_cvt_kernel<<<dim3(H_ / 32, D_ / 32, E_), dim3(32, 8), 0, stream>>>(W1, W1T, D_, H_);
    moe_transpose_cvt_kernel<<<dim3(O_ / 32, H_ / 32, E_), dim3(32, 8), 0, stream>>>(W2, W2T, H_, O_);
    moe_gate_kernel<<<N_ / 8, 256, 0, stream>>>(x, Wg, bg, counts, toklist, gatew);

    for (int e = 0; e < E_; e++) {
        moe_gemm1_kernel<<<dim3(H_ / 128, N_ / 128), 256, 0, stream>>>(
            xb, W1T, b1, counts, toklist, hbuf, e);
        moe_gemm2_kernel<<<dim3(O_ / 128, N_ / 128), 256, 0, stream>>>(
            hbuf, W2T, b2, counts, toklist, gatew, out, e);
    }
}
